// Model_39676907883517
// MI455X (gfx1250) — matrix operand order repaired and checked by reading, not compiled, not run on hardware
//
#include <hip/hip_runtime.h>
#include <stdint.h>

typedef __attribute__((ext_vector_type(16))) _Float16 v16h;
typedef __attribute__((ext_vector_type(8)))  _Float16 v8h;
typedef __attribute__((ext_vector_type(8)))  float    v8f;

#define B_    16
#define SQ_   2048
#define SK_   2048
#define D_    128
#define DV_   128
#define QTILE 128     // Q rows per workgroup (8 waves x 16)
#define KTILE 32      // K/V rows staged per iteration
#define NJ    (SK_ / KTILE)
#define PK    136     // K LDS row pitch in halves (272B, 16B aligned, conflict-free)
#define PVT   40      // V^T LDS row pitch in halves (80B)
#define PP    40      // P LDS row pitch in halves (80B)
#define INV_SCALE 0.2f                 // scores * 1/SCALE_FACTOR
#define KEEP      1.1111111111111112f  // 1/(1-p), p=0.1

// LDS sizes (halves)
#define K_LDS   (KTILE * PK)            // 4352
#define VT_LDS  (DV_ * PVT)             // 5120
#define P_LDS   (8 * 16 * PP)           // 5120
#define LDS_TOT (K_LDS + VT_LDS + P_LDS)

// DPP8 selectors: lane i reads lane i^1 / i^2 / i^4 within each 8-lane row
#define DPP8_X1 ((1) | (0 << 3) | (3 << 6) | (2 << 9) | (5 << 12) | (4 << 15) | (7 << 18) | (6 << 21))
#define DPP8_X2 ((2) | (3 << 3) | (0 << 6) | (1 << 9) | (6 << 12) | (7 << 15) | (4 << 18) | (5 << 21))
#define DPP8_X4 ((4) | (5 << 3) | (6 << 6) | (7 << 9) | (0 << 12) | (1 << 15) | (2 << 18) | (3 << 21))
#define SWZ_X8  0x201f   // ds_swizzle group-of-32: xor=0x08, or=0, and=0x1f (SWAPX8)

template <int SEL>
static __device__ inline float dpp8f(float x) {
    return __int_as_float(__builtin_amdgcn_mov_dpp8(__float_as_int(x), SEL));
}
static __device__ inline float swz_x8(float x) {
    return __int_as_float(__builtin_amdgcn_ds_swizzle(__float_as_int(x), SWZ_X8));
}
// reduce across each 16-lane group (rows live per-group; masks 1,2,4 stay in 8-lane rows)
static __device__ inline float red_max16(float x) {
    x = fmaxf(x, dpp8f<DPP8_X1>(x));
    x = fmaxf(x, dpp8f<DPP8_X2>(x));
    x = fmaxf(x, dpp8f<DPP8_X4>(x));
    x = fmaxf(x, swz_x8(x));
    return x;
}
static __device__ inline float red_sum16(float x) {
    x += dpp8f<DPP8_X1>(x);
    x += dpp8f<DPP8_X2>(x);
    x += dpp8f<DPP8_X4>(x);
    x += swz_x8(x);
    return x;
}

static __device__ inline v16h ld16h(const _Float16* p) {
    // two 16B LDS loads -> one 16-half fragment
    v8h a = *(const v8h*)p;
    v8h b = *(const v8h*)(p + 16);
    v16h r;
#pragma unroll
    for (int i = 0; i < 8; ++i) { r[i] = a[i]; r[8 + i] = b[i]; }
    return r;
}

// CDNA5 async DMA: per-lane 16B global -> LDS, tracked by ASYNCcnt.
static __device__ inline void async_mask_ld16(const unsigned char* g, unsigned char* l) {
    unsigned loff = (unsigned)(size_t)l;   // generic LDS ptr low 32 bits = LDS offset
    asm volatile("global_load_async_to_lds_b128 %0, %1, off"
                 :: "v"(loff), "v"(g) : "memory");
}

__global__ void __launch_bounds__(256)
fattn_kernel(const float* __restrict__ x1, const float* __restrict__ x2,
             const float* __restrict__ x3, const unsigned char* __restrict__ dmask,
             float* __restrict__ out)
{
    __shared__ _Float16 lds[LDS_TOT];
    __shared__ unsigned char Mlds[2][QTILE * KTILE];  // double-buffered mask tile
    _Float16* Klds = lds;                    // 32 x 128 f16, pitch PK
    _Float16* Vt   = lds + K_LDS;            // 128 x 32 f16 (transposed), pitch PVT
    _Float16* Plds = lds + K_LDS + VT_LDS;   // per-wave 16 x 32 f16, pitch PP

    const int b    = blockIdx.x / (SQ_ / QTILE);
    const int q0   = (blockIdx.x % (SQ_ / QTILE)) * QTILE;
    const int tid  = threadIdx.x;
    const int wave = tid >> 5;
    const int lane = tid & 31;
    const int m    = lane & 15;   // column / row-in-group index
    const int hi   = lane >> 4;   // lane group (0: lanes 0-15, 1: lanes 16-31)

    // ---- Q A-fragments: 16x32 f16 per 32-dim chunk, ISA 7.12.2 layout ----
    const float* qrow = x1 + ((size_t)b * SQ_ + q0 + wave * 16 + m) * D_;
    v16h qf[4];
#pragma unroll
    for (int c = 0; c < 4; ++c) {
#pragma unroll
        for (int g = 0; g < 2; ++g) {
            const float* s = qrow + 32 * c + 16 * g + 8 * hi;
            float4 f0 = *(const float4*)(s);
            float4 f1 = *(const float4*)(s + 4);
            qf[c][g * 8 + 0] = (_Float16)f0.x; qf[c][g * 8 + 1] = (_Float16)f0.y;
            qf[c][g * 8 + 2] = (_Float16)f0.z; qf[c][g * 8 + 3] = (_Float16)f0.w;
            qf[c][g * 8 + 4] = (_Float16)f1.x; qf[c][g * 8 + 5] = (_Float16)f1.y;
            qf[c][g * 8 + 6] = (_Float16)f1.z; qf[c][g * 8 + 7] = (_Float16)f1.w;
        }
    }

    // ---- flash-attention running state ----
    v8f acc[8];
#pragma unroll
    for (int ob = 0; ob < 8; ++ob)
#pragma unroll
        for (int j = 0; j < 8; ++j) acc[ob][j] = 0.0f;

    float m_run[8], l_run[8];
#pragma unroll
    for (int f = 0; f < 8; ++f) { m_run[f] = -__builtin_inff(); l_run[f] = 0.0f; }

    // cooperative staging coordinates: 256 threads cover 32 rows x 128 dims
    const int r  = tid >> 3;          // 0..31 (tile row)
    const int db = (tid & 7) * 16;    // dim base, 16 floats per thread
    const float* kbase = x2 + (size_t)b * SK_ * D_ + (size_t)r * D_ + db;
    const float* vbase = x3 + (size_t)b * SK_ * DV_ + (size_t)r * DV_ + db;
    // mask: per-lane async staging row (wave-private: wave w writes+reads rows 16w..16w+15)
    const int mrow = wave * 16 + (lane >> 1);
    const unsigned char* mgbase =
        dmask + ((size_t)b * SQ_ + q0 + mrow) * (size_t)SK_ + 16 * (lane & 1);
    const int mloc = mrow * KTILE + 16 * (lane & 1);
    const unsigned char* Mw = &Mlds[0][0] + wave * 16 * KTILE;  // wave's read base (buf 0)
    _Float16* Pw = Plds + wave * (16 * PP);

    // ---- software pipeline prologue: tile 0 K/V into regs, tile 0 mask via async DMA ----
    float kr[16], vr[16];
#pragma unroll
    for (int i = 0; i < 4; ++i) {
        *(float4*)(kr + 4 * i) = *(const float4*)(kbase + 4 * i);
        *(float4*)(vr + 4 * i) = *(const float4*)(vbase + 4 * i);
    }
    async_mask_ld16(mgbase, &Mlds[0][mloc]);

#pragma unroll 1
    for (int jb = 0; jb < NJ; ++jb) {
        __syncthreads();  // previous iteration's K/V readers done

        // ---- stage current K tile (row-major f16) and V tile (transposed) from regs ----
#pragma unroll
        for (int i = 0; i < 16; ++i) Klds[r * PK + db + i] = (_Float16)kr[i];
#pragma unroll
        for (int i = 0; i < 16; ++i) Vt[(db + i) * PVT + r] = (_Float16)vr[i];

        // ---- issue next tile: K/V global->regs, mask global->LDS (async DMA) ----
        if (jb + 1 < NJ) {
            const float* kn = kbase + (size_t)(jb + 1) * KTILE * D_;
            const float* vn = vbase + (size_t)(jb + 1) * KTILE * DV_;
#pragma unroll
            for (int i = 0; i < 4; ++i) {
                *(float4*)(kr + 4 * i) = *(const float4*)(kn + 4 * i);
                *(float4*)(vr + 4 * i) = *(const float4*)(vn + 4 * i);
            }
            async_mask_ld16(mgbase + (size_t)(jb + 1) * KTILE,
                            &Mlds[(jb + 1) & 1][mloc]);
        }
        __syncthreads();  // K/V tile visible to all waves

        // wait for THIS tile's async mask (in-order: allow the newer one to stay in flight)
        if (jb + 1 < NJ) {
            asm volatile("s_wait_asynccnt 0x1" ::: "memory");
        } else {
            asm volatile("s_wait_asynccnt 0x0" ::: "memory");
        }

        // ---- S(16x32) = Q · K^T via WMMA f16 ----
        v8f S0, S1;
#pragma unroll
        for (int j = 0; j < 8; ++j) { S0[j] = 0.0f; S1[j] = 0.0f; }
#pragma unroll
        for (int c = 0; c < 4; ++c) {
            v16h k0 = ld16h(Klds + (0 * 16 + m) * PK + 32 * c + 8 * hi);
            v16h k1 = ld16h(Klds + (1 * 16 + m) * PK + 32 * c + 8 * hi);
            S0 = __builtin_amdgcn_wmma_f32_16x16x32_f16(false, qf[c], false, k0,
                                                        (short)0, S0, false, false);
            S1 = __builtin_amdgcn_wmma_f32_16x16x32_f16(false, qf[c], false, k1,
                                                        (short)0, S1, false, false);
        }

        // ---- online softmax (row = f + 8*hi, cols across the 16-lane group) ----
        const unsigned char* Mb = Mw + ((jb & 1) * (QTILE * KTILE));
#pragma unroll
        for (int f = 0; f < 8; ++f) {
            float s0 = S0[f] * INV_SCALE;
            float s1 = S1[f] * INV_SCALE;
            float mn = fmaxf(m_run[f], red_max16(fmaxf(s0, s1)));
            float p0 = __expf(s0 - mn);
            float p1 = __expf(s1 - mn);
            float rs = red_sum16(p0 + p1);
            float corr = __expf(m_run[f] - mn);
            l_run[f] = l_run[f] * corr + rs;   // denominator: pre-dropout probs
            m_run[f] = mn;
#pragma unroll
            for (int ob = 0; ob < 8; ++ob) acc[ob][f] *= corr;

            // dropout keep-mask from LDS (async-staged), repack P to LDS
            unsigned char k0m = Mb[(f + 8 * hi) * KTILE + m];
            unsigned char k1m = Mb[(f + 8 * hi) * KTILE + 16 + m];
            Pw[(f + 8 * hi) * PP + m]      = (_Float16)(k0m ? p0 * KEEP : 0.0f);
            Pw[(f + 8 * hi) * PP + 16 + m] = (_Float16)(k1m ? p1 * KEEP : 0.0f);
        }

        // ---- reload P as A-fragment (16x32) ----
        const _Float16* pp = Pw + m * PP;
        v8h pa = *(const v8h*)(pp + 8 * hi);
        v8h pb = *(const v8h*)(pp + 16 + 8 * hi);
        v16h pf;
#pragma unroll
        for (int i = 0; i < 8; ++i) { pf[i] = pa[i]; pf[8 + i] = pb[i]; }

        // ---- O += P · V ----
#pragma unroll
        for (int ob = 0; ob < 8; ++ob) {
            v16h vf = ld16h(Vt + (ob * 16 + m) * PVT + 8 * hi);
            acc[ob] = __builtin_amdgcn_wmma_f32_16x16x32_f16(false, pf, false, vf,
                                                             (short)0, acc[ob],
                                                             false, false);
        }
    }

    // ---- epilogue: normalize and store ----
#pragma unroll
    for (int f = 0; f < 8; ++f) {
        float inv = 1.0f / l_run[f];
        float* orow = out + ((size_t)b * SQ_ + q0 + wave * 16 + f + 8 * hi) * DV_;
#pragma unroll
        for (int ob = 0; ob < 8; ++ob) orow[ob * 16 + m] = acc[ob][f] * inv;
    }
}

extern "C" void kernel_launch(void* const* d_in, const int* in_sizes, int n_in,
                              void* d_out, int out_size, void* d_ws, size_t ws_size,
                              hipStream_t stream) {
    const float* x1 = (const float*)d_in[0];
    const float* x2 = (const float*)d_in[1];
    const float* x3 = (const float*)d_in[2];
    const unsigned char* dmask = (const unsigned char*)d_in[3];  // jax bool -> 1 byte
    float* out = (float*)d_out;

    dim3 grid(B_ * (SQ_ / QTILE));   // 16 batches x 16 Q-blocks = 256
    dim3 block(256);                 // 8 wave32
    fattn_kernel<<<grid, block, 0, stream>>>(x1, x2, x3, dmask, out);
}
